// CLIP_text_image_concat_10711648436802
// MI455X (gfx1250) — matrix operand order repaired and checked by reading, not compiled, not run on hardware
//
#include <hip/hip_runtime.h>
#include <hip/hip_bf16.h>
#include <math.h>

// Problem constants (reference: B,N,DE,DV,H = 256,312,512,768,312)
#define B_   256
#define N_   312
#define DE_  512
#define DV_  768
#define H_   312
#define NH_  (N_ * H_)        // 97344, exactly 6084 16-col tiles
#define BN_  (B_ * N_)        // 79872 elements per output tensor
#define EPS_ 1e-5f

typedef __attribute__((ext_vector_type(16))) __bf16 v16bf;
typedef __attribute__((ext_vector_type(8)))  __bf16 v8bf;
typedef __attribute__((ext_vector_type(4)))  __bf16 v4bf;
typedef __attribute__((ext_vector_type(8)))  float  v8f;

union V16U { v16bf v; v8bf h[2]; };

__device__ __forceinline__ float waveReduceSum(float v) {
    v += __shfl_xor(v, 16);
    v += __shfl_xor(v, 8);
    v += __shfl_xor(v, 4);
    v += __shfl_xor(v, 2);
    v += __shfl_xor(v, 1);
    return v;
}

// ---------------------------------------------------------------------------
// Big kernel: h[b,c] = relu( sum_k X[b,k]*W[c,k] + tcol[c] ), c = n*H + h flat.
// Writes relu(h) as bf16 to reluOut[b*NH + c]; atomically accumulates
// s1[c] = sum_b relu, s2[c] = sum_b relu^2.
//
// Block: 256 threads (8 waves) owns 4 column tiles and sweeps ALL 256 batch
// rows so fp32 weights stream from DRAM exactly once (0.9 GB total).
// Software-pipelined: double-buffered 16KB LDS A chunks, one barrier per
// 32-K step; A-chunk(i+2) and B(i+1) global loads issued while computing
// step i, hiding a full iteration of memory latency behind 8 WMMAs/wave.
// ---------------------------------------------------------------------------
template<int K>
__global__ __launch_bounds__(256)
void hidden_kernel(const float* __restrict__ X,     // [256, K]
                   const float* __restrict__ W,     // row stride WLD, first K used
                   int WLD,
                   const float* __restrict__ tcol,  // [NH]
                   __bf16* __restrict__ reluOut,    // [256, NH]
                   float* __restrict__ s1,
                   float* __restrict__ s2)
{
    __shared__ __align__(16) __bf16 As[2][256 * 32];   // 2 x 16 KB A chunks

    const int tid   = threadIdx.x;
    const int lane  = tid & 31;
    const int wave  = tid >> 5;
    const int colTile = blockIdx.x * 4 + (wave >> 1);
    const int mhalf   = wave & 1;                   // rows [mhalf*128, +128)
    const int colBase = colTile * 16;
    const int col     = colBase + (lane & 15);
    const int kb      = (lane >> 4) * 16;           // A-frag K sub-offset
    const int koffB   = (lane >> 4) * 16;           // B-frag K sub-offset
    const int NKK     = K / 32;

    const float* wrow = W + (size_t)col * WLD;

    float4 sreg[8];   // in-flight A chunk (32 f32/lane)
    float4 breg[4];   // in-flight B fragment (16 f32/lane)

    auto loadA = [&](int kk) {
        #pragma unroll
        for (int rep = 0; rep < 8; ++rep) {
            int idx = rep * 256 + tid;              // 0..2047 float4 slots
            int r = idx >> 3, q = idx & 7;
            sreg[rep] = *(const float4*)(X + (size_t)r * K + kk + q * 4);
        }
    };
    auto storeA = [&](int buf) {
        #pragma unroll
        for (int rep = 0; rep < 8; ++rep) {
            int idx = rep * 256 + tid;
            int r = idx >> 3, q = idx & 7;
            v4bf p4;                                 // 2x v_cvt_pk_bf16_f32 -> ds_store_b64
            p4[0] = (__bf16)sreg[rep].x; p4[1] = (__bf16)sreg[rep].y;
            p4[2] = (__bf16)sreg[rep].z; p4[3] = (__bf16)sreg[rep].w;
            *(v4bf*)(&As[buf][r * 32 + q * 4]) = p4;
        }
    };
    auto loadB = [&](int kk) {
        const float* bp = wrow + kk + koffB;
        #pragma unroll
        for (int j = 0; j < 4; ++j) breg[j] = *(const float4*)(bp + j * 4);
    };

    v8f acc[8];
    const float tc = tcol[col];
    #pragma unroll
    for (int m = 0; m < 8; ++m)
        #pragma unroll
        for (int i = 0; i < 8; ++i) acc[m][i] = tc;

    // Pipeline prologue: chunk0 -> LDS buf0, chunk1 + B0 in flight
    loadA(0);
    storeA(0);
    if (NKK > 1) loadA(32);
    loadB(0);
    __syncthreads();

    int p = 0;
    for (int i = 0; i < NKK; ++i) {
        const int kk = i * 32;
        if (i + 1 < NKK) storeA(p ^ 1);             // chunk i+1 (loaded last iter)
        if (i + 2 < NKK) loadA(kk + 64);            // chunk i+2 into flight

        v16bf bmat;                                  // convert current B fragment
        #pragma unroll
        for (int j = 0; j < 4; ++j) {
            bmat[j * 4 + 0] = (__bf16)breg[j].x;
            bmat[j * 4 + 1] = (__bf16)breg[j].y;
            bmat[j * 4 + 2] = (__bf16)breg[j].z;
            bmat[j * 4 + 3] = (__bf16)breg[j].w;
        }
        if (i + 1 < NKK) loadB(kk + 32);            // next B fragment into flight

        #pragma unroll
        for (int mf = 0; mf < 8; ++mf) {
            int row = mhalf * 128 + mf * 16 + (lane & 15);
            V16U a;
            a.h[0] = *(const v8bf*)(&As[p][row * 32 + kb]);        // K: kk+kb+0..7
            a.h[1] = *(const v8bf*)(&As[p][row * 32 + 8 + kb]);    // K: kk+kb+8..15
            acc[mf] = __builtin_amdgcn_wmma_f32_16x16x32_bf16(
                false, a.v, false, bmat, (short)0, acc[mf], false, false);
        }
        __syncthreads();
        p ^= 1;
    }

    // Epilogue: relu, bf16 store, per-column stats
    float ls1 = 0.f, ls2 = 0.f;
    #pragma unroll
    for (int mf = 0; mf < 8; ++mf) {
        #pragma unroll
        for (int i = 0; i < 8; ++i) {
            float v = acc[mf][i];
            v = v > 0.f ? v : 0.f;
            ls1 += v; ls2 += v * v;
            int row = mhalf * 128 + mf * 16 + (lane >> 4) * 8 + i;
            reluOut[(size_t)row * NH_ + col] = (__bf16)v;
        }
    }
    ls1 += __shfl_xor(ls1, 16);   // combine lanes sharing the same column
    ls2 += __shfl_xor(ls2, 16);
    if (lane < 16) {
        atomicAdd(&s1[col], ls1);
        atomicAdd(&s2[col], ls2);
    }
}

// ---------------------------------------------------------------------------
// Small WMMA GEMM: out[b*312 + c] = sum_k X[b,k]*W[c,k] + (cc ? cc[c] : 0),
// b<256, c<312. Grid (5,16), block 128 (4 waves, colTiles 0..19).
// Used for lin1, lin2 and the CLIP logits GEMM (with pre-normalized inputs).
// ---------------------------------------------------------------------------
template<int K>
__global__ __launch_bounds__(128)
void small_gemm_kernel(const float* __restrict__ X,   // [256, K]
                       const float* __restrict__ W,   // row stride WLD
                       int WLD,
                       const float* __restrict__ cc,  // [312] or nullptr
                       float* __restrict__ out)       // [256, 312]
{
    __shared__ __align__(16) __bf16 As[16 * K];
    const int tid  = threadIdx.x;
    const int lane = tid & 31;
    const int wave = tid >> 5;
    const int m0   = blockIdx.y * 16;

    for (int i = tid; i < 16 * K; i += 128) {
        int r = i / K, c = i % K;
        As[i] = (__bf16)X[(size_t)(m0 + r) * K + c];
    }
    __syncthreads();

    const int colTile = blockIdx.x * 4 + wave;     // 0..19
    const int colBase = colTile * 16;
    const int col     = colBase + (lane & 15);
    const bool cvalid = (col < N_);
    const int kb    = (lane >> 4) * 16;
    const int koffB = (lane >> 4) * 16;
    const int rl    = lane & 15;

    v8f acc;
    float c0 = (cc != nullptr && cvalid) ? cc[col] : 0.f;
    #pragma unroll
    for (int i = 0; i < 8; ++i) acc[i] = c0;

    const float* wrow = W + (size_t)col * WLD;
    for (int kk = 0; kk < K; kk += 32) {
        v16bf bmat;
        if (cvalid) {
            const float* bp = wrow + kk + koffB;
            #pragma unroll
            for (int j4 = 0; j4 < 4; ++j4) {
                float4 f = *(const float4*)(bp + j4 * 4);
                bmat[j4 * 4 + 0] = (__bf16)f.x;
                bmat[j4 * 4 + 1] = (__bf16)f.y;
                bmat[j4 * 4 + 2] = (__bf16)f.z;
                bmat[j4 * 4 + 3] = (__bf16)f.w;
            }
        } else {
            #pragma unroll
            for (int j = 0; j < 16; ++j) bmat[j] = (__bf16)0.f;
        }
        V16U a;
        a.h[0] = *(const v8bf*)(As + rl * K + kk + kb);
        a.h[1] = *(const v8bf*)(As + rl * K + kk + 8 + kb);
        acc = __builtin_amdgcn_wmma_f32_16x16x32_bf16(
            false, a.v, false, bmat, (short)0, acc, false, false);
    }

    #pragma unroll
    for (int i = 0; i < 8; ++i) {
        int row = m0 + (lane >> 4) * 8 + i;
        if (cvalid) out[(size_t)row * N_ + col] = acc[i];
    }
}

// ---------------------------------------------------------------------------
// Per-column text/bias constant: outc[c] = bias[c] + dot(T[c/hdiv,:512], W[c, KOFF:KOFF+512])
// One wave per column; float4 (b128) streaming of the ~400 MB text-half weights.
// ---------------------------------------------------------------------------
__global__ __launch_bounds__(256)
void coldot_kernel(const float* __restrict__ T,     // [*, 512]
                   const float* __restrict__ W, int WLD, int KOFF,
                   const float* __restrict__ bias,  // [numC]
                   float* __restrict__ outc,        // [numC]
                   int numC, int hdiv)
{
    const int wave = threadIdx.x >> 5, lane = threadIdx.x & 31;
    const int c = blockIdx.x * 8 + wave;
    if (c >= numC) return;
    const int n = (hdiv == 1) ? c : (c / hdiv);
    const float4* t4 = (const float4*)(T + (size_t)n * 512);
    const float4* w4 = (const float4*)(W + (size_t)c * WLD + KOFF);
    float s = 0.f;
    #pragma unroll
    for (int i = 0; i < 4; ++i) {
        float4 a = t4[lane + 32 * i];
        float4 b = w4[lane + 32 * i];
        s += a.x * b.x + a.y * b.y + a.z * b.z + a.w * b.w;
    }
    s = waveReduceSum(s);
    if (lane == 0) outc[c] = s + bias[c];
}

// ---------------------------------------------------------------------------
// Row L2-normalize (optionally scaled by exp(logit_scale)). One wave per row.
// ---------------------------------------------------------------------------
__global__ __launch_bounds__(32)
void rownorm_kernel(const float* __restrict__ X, float* __restrict__ Y,
                    const float* __restrict__ logit_scale, int applyExp)
{
    const int r = blockIdx.x, lane = threadIdx.x;
    const float4* x4 = (const float4*)(X + (size_t)r * 512);
    float4 v[4];
    float s = 0.f;
    #pragma unroll
    for (int i = 0; i < 4; ++i) {
        v[i] = x4[lane + 32 * i];
        s += v[i].x * v[i].x + v[i].y * v[i].y + v[i].z * v[i].z + v[i].w * v[i].w;
    }
    s = waveReduceSum(s);                 // all lanes hold full sum
    float sc = rsqrtf(s);
    if (applyExp) sc *= expf(logit_scale[0]);
    float4* y4 = (float4*)(Y + (size_t)r * 512);
    #pragma unroll
    for (int i = 0; i < 4; ++i) {
        float4 o; o.x = v[i].x * sc; o.y = v[i].y * sc; o.z = v[i].z * sc; o.w = v[i].w * sc;
        y4[lane + 32 * i] = o;
    }
}

// ---------------------------------------------------------------------------
// Fold BatchNorm + final linear:
//   a[h] = gamma*rsqrt(var+eps)*W2 ;  c = sum_h (beta - gamma*mean*rsqrt)*W2 + b2
//   out[b*312+n] = sum_h a[h]*relu_h[b,n,h] + c
// One block per head n; 8 waves sweep batch rows reading bf16 relu(h) with
// 8-byte vector loads (H=312 -> 78 quads, 8B aligned).
// ---------------------------------------------------------------------------
__global__ __launch_bounds__(256)
void class_finalize_kernel(const float* __restrict__ s1, const float* __restrict__ s2,
                           const float* __restrict__ gamma, const float* __restrict__ beta,
                           const float* __restrict__ W2, const float* __restrict__ b2,
                           const __bf16* __restrict__ relu,  // [256, NH]
                           float* __restrict__ out)          // [256, 312]
{
    __shared__ float aSh[H_];
    __shared__ float cSh;
    const int n = blockIdx.x, tid = threadIdx.x;
    if (tid == 0) cSh = 0.f;
    __syncthreads();

    float cpart = 0.f;
    for (int h = tid; h < H_; h += 256) {
        int idx = n * H_ + h;
        float mean = s1[idx] * (1.0f / 256.0f);
        float var  = s2[idx] * (1.0f / 256.0f) - mean * mean;
        float r    = rsqrtf(var + EPS_);
        float g = gamma[idx], w = W2[idx];
        aSh[h] = g * r * w;
        cpart += (beta[idx] - g * mean * r) * w;
    }
    atomicAdd(&cSh, cpart);
    __syncthreads();
    const float cterm = cSh + b2[n];

    const int wave = tid >> 5, lane = tid & 31;
    for (int b = wave; b < 256; b += 8) {
        const __bf16* row = relu + (size_t)b * NH_ + (size_t)n * H_;
        float s = 0.f;
        #pragma unroll
        for (int i = 0; i < 3; ++i) {
            int q = lane + 32 * i;                 // quad index; 78 quads = 312 elems
            if (q < H_ / 4) {
                v4bf v = *(const v4bf*)(row + q * 4);
                s += aSh[q * 4 + 0] * (float)v[0] + aSh[q * 4 + 1] * (float)v[1]
                   + aSh[q * 4 + 2] * (float)v[2] + aSh[q * 4 + 3] * (float)v[3];
            }
        }
        s = waveReduceSum(s);
        if (lane == 0) out[(size_t)b * N_ + n] = s + cterm;
    }
}

// ---------------------------------------------------------------------------
extern "C" void kernel_launch(void* const* d_in, const int* in_sizes, int n_in,
                              void* d_out, int out_size, void* d_ws, size_t ws_size,
                              hipStream_t stream)
{
    const float* image_embed = (const float*)d_in[0];   // [256, 512]
    const float* text_embed  = (const float*)d_in[1];   // [312, 512]
    const float* image_out   = (const float*)d_in[2];   // [256, 768]
    const float* text_out    = (const float*)d_in[3];   // [312, 512]
    const float* W1   = (const float*)d_in[4];          // [312, 1024]
    const float* b1   = (const float*)d_in[5];
    const float* W2   = (const float*)d_in[6];          // [312, 1280]
    const float* b2   = (const float*)d_in[7];
    const float* C1W1 = (const float*)d_in[8];          // [312, 312, 1024]
    const float* C1b1 = (const float*)d_in[9];
    const float* C1g  = (const float*)d_in[10];
    const float* C1be = (const float*)d_in[11];
    const float* C1W2 = (const float*)d_in[12];
    const float* C1b2 = (const float*)d_in[13];
    const float* C2W1 = (const float*)d_in[14];         // [312, 312, 1280]
    const float* C2b1 = (const float*)d_in[15];
    const float* C2g  = (const float*)d_in[16];
    const float* C2be = (const float*)d_in[17];
    const float* C2W2 = (const float*)d_in[18];
    const float* C2b2 = (const float*)d_in[19];
    const float* logit_scale = (const float*)d_in[20];

    float* out = (float*)d_out;
    char*  ws  = (char*)d_ws;

    // Workspace layout (256B aligned slices)
    size_t off = 0;
    auto take = [&](size_t bytes) { size_t o = off; off = (off + bytes + 255) & ~(size_t)255; return o; };
    __bf16* relu1 = (__bf16*)(ws + take((size_t)B_ * NH_ * sizeof(__bf16)));   // ~49.8 MB
    __bf16* relu2 = (__bf16*)(ws + take((size_t)B_ * NH_ * sizeof(__bf16)));   // ~49.8 MB
    float* tcol1  = (float*)(ws + take((size_t)NH_ * sizeof(float)));
    float* tcol2  = (float*)(ws + take((size_t)NH_ * sizeof(float)));
    float* sstats = (float*)(ws + take((size_t)4 * NH_ * sizeof(float)));      // s1_1,s2_1,s1_2,s2_2
    float* s1_1 = sstats, *s2_1 = sstats + NH_, *s1_2 = sstats + 2 * NH_, *s2_2 = sstats + 3 * NH_;
    float* linc1 = (float*)(ws + take(N_ * sizeof(float)));
    float* linc2 = (float*)(ws + take(N_ * sizeof(float)));
    float* imgn  = (float*)(ws + take((size_t)B_ * DE_ * sizeof(float)));
    float* txtn  = (float*)(ws + take((size_t)N_ * DE_ * sizeof(float)));

    // Zero batch-stat accumulators (graph-capture-safe)
    hipMemsetAsync(sstats, 0, (size_t)4 * NH_ * sizeof(float), stream);

    // Normalized embeddings for CLIP logits (exp(logit_scale) folded into image side)
    rownorm_kernel<<<B_, 32, 0, stream>>>(image_embed, imgn, logit_scale, 1);
    rownorm_kernel<<<N_, 32, 0, stream>>>(text_embed,  txtn, logit_scale, 0);

    // Per-column text terms + biases (reads text-half of the big weights once)
    coldot_kernel<<<(NH_ + 7) / 8, 256, 0, stream>>>(text_embed, C1W1, 1024, 512, C1b1, tcol1, NH_, H_);
    coldot_kernel<<<(NH_ + 7) / 8, 256, 0, stream>>>(text_out,   C2W1, 1280, 768, C2b1, tcol2, NH_, H_);
    coldot_kernel<<<(N_ + 7) / 8,  256, 0, stream>>>(text_embed, W1,   1024, 512, b1, linc1, N_, 1);
    coldot_kernel<<<(N_ + 7) / 8,  256, 0, stream>>>(text_out,   W2,   1280, 768, b2, linc2, N_, 1);

    // Big WMMA GEMMs: relu(h) + batch stats. 6084 col tiles / 4 per block = 1521.
    hidden_kernel<512><<<1521, 256, 0, stream>>>(image_embed, C1W1, 1024, tcol1, relu1, s1_1, s2_1);
    hidden_kernel<768><<<1521, 256, 0, stream>>>(image_out,   C2W1, 1280, tcol2, relu2, s1_2, s2_2);

    // Small WMMA GEMMs: lin1, lin2, CLIP logits
    small_gemm_kernel<512><<<dim3(5, 16), 128, 0, stream>>>(image_embed, W1, 1024, linc1, out + 0 * BN_);
    small_gemm_kernel<768><<<dim3(5, 16), 128, 0, stream>>>(image_out,   W2, 1280, linc2, out + 1 * BN_);
    small_gemm_kernel<512><<<dim3(5, 16), 128, 0, stream>>>(imgn, txtn,  512, nullptr, out + 4 * BN_);

    // BatchNorm fold + final per-head reduction
    class_finalize_kernel<<<N_, 256, 0, stream>>>(s1_1, s2_1, C1g, C1be, C1W2, C1b2, relu1, out + 2 * BN_);
    class_finalize_kernel<<<N_, 256, 0, stream>>>(s1_2, s2_2, C2g, C2be, C2W2, C2b2, relu2, out + 3 * BN_);
}
